// LSTM_23905787969736
// MI455X (gfx1250) — matrix operand order repaired and checked by reading, not compiled, not run on hardware
//
#include <hip/hip_runtime.h>
#include <hip/hip_bf16.h>
#include <stdint.h>

typedef __bf16 bf16_t;
typedef __attribute__((ext_vector_type(16))) __bf16 v16bf;
typedef __attribute__((ext_vector_type(8)))  __bf16 v8bf;
typedef __attribute__((ext_vector_type(8)))  float  v8f;
typedef __attribute__((ext_vector_type(4)))  unsigned int u32x4;
typedef __attribute__((ext_vector_type(8)))  int i32x8;
typedef __attribute__((ext_vector_type(4)))  int i32x4;

static constexpr int BATCH = 128, SEQ = 256, IN = 1024, HID = 1024;
static constexpr int KTOT = IN + HID;          // 2048

// ---- workspace layout (bytes) ----
static constexpr size_t OFF_WH  = 0;                                   // bf16 [4][H][1024] (h-part of W)
static constexpr size_t SZ_WH   = (size_t)4 * HID * 1024 * 2;          // 8 MB
static constexpr size_t OFF_WX  = OFF_WH + SZ_WH;                      // bf16 [4][H][1024] (x-part of W)
static constexpr size_t SZ_WX   = SZ_WH;
static constexpr size_t OFF_XBF = OFF_WX + SZ_WX;                      // bf16 [B][S][I]
static constexpr size_t SZ_XBF  = (size_t)BATCH * SEQ * IN * 2;        // 67 MB
static constexpr size_t OFF_HBF = OFF_XBF + SZ_XBF;                    // bf16 [B][H]
static constexpr size_t SZ_HBF  = (size_t)BATCH * HID * 2;
static constexpr size_t OFF_CNT = OFF_HBF + SZ_HBF;                    // u32 [SEQ] grid-barrier counters

// ---- LDS layout of the scan kernel (dynamic shared) ----
// W_h slice: 4 gates x 32 outputs, each row 1024 bf16 + TDM pad 16B per 1024B -> 2080 B/row
static constexpr int LDS_WH_ROW   = 2080;
static constexpr int LDS_WH_BYTES = 4 * 32 * LDS_WH_ROW;               // 266240
static constexpr int LDS_BIAS_OFF = LDS_WH_BYTES;                      // float [4][32]
static constexpr int LDS_GATE_OFF = LDS_BIAS_OFF + 512;                // float [4][16][32]
static constexpr int LDS_C_OFF    = LDS_GATE_OFF + 8192;               // float [16][32]
static constexpr int SMEM_BYTES   = LDS_C_OFF + 2048;                  // 276992 < 320KB

static __device__ __forceinline__ float sigm(float x)  { return 1.f / (1.f + __expf(-x)); }
static __device__ __forceinline__ float tanha(float x) { return 2.f / (1.f + __expf(-2.f * x)) - 1.f; }

// ------------------------------------------------------------------
// Prep: fp32 -> bf16 weight split (hidden part | input part)
// ------------------------------------------------------------------
__global__ __launch_bounds__(256) void pack_weights(
    const float* __restrict__ Wf, const float* __restrict__ Wi,
    const float* __restrict__ Wc, const float* __restrict__ Wo,
    bf16_t* __restrict__ wh, bf16_t* __restrict__ wx)
{
    const float* Ws[4] = {Wf, Wi, Wc, Wo};
    const size_t per_g = (size_t)HID * KTOT;
    const size_t total = 4 * per_g;
    for (size_t i = (size_t)blockIdx.x * blockDim.x + threadIdx.x; i < total;
         i += (size_t)gridDim.x * blockDim.x) {
        int    g   = (int)(i / per_g);
        size_t rem = i % per_g;
        int    c   = (int)(rem % KTOT);
        int    r   = (int)(rem / KTOT);
        float  v   = Ws[g][rem];
        if (c < HID) wh[((size_t)g * HID + r) * 1024 + c]         = (bf16_t)v;
        else         wx[((size_t)g * HID + r) * 1024 + (c - HID)] = (bf16_t)v;
    }
}

__global__ __launch_bounds__(256) void pack_x(const float* __restrict__ x, bf16_t* __restrict__ xbf)
{
    const size_t total = (size_t)BATCH * SEQ * IN;
    for (size_t i = (size_t)blockIdx.x * blockDim.x + threadIdx.x; i < total;
         i += (size_t)gridDim.x * blockDim.x)
        xbf[i] = (bf16_t)x[i];
}

__global__ __launch_bounds__(256) void init_state(unsigned int* __restrict__ hbf_u32,
                                                  unsigned int* __restrict__ cnt)
{
    int i = blockIdx.x * blockDim.x + threadIdx.x;
    if (i < BATCH * HID / 2) hbf_u32[i] = 0u;   // h0 = 0 (bf16 pairs)
    if (i < SEQ)             cnt[i]     = 0u;   // grid-barrier generations
}

// ------------------------------------------------------------------
// Persistent LSTM scan.
// grid = 8 batch-chunks (M=16) x 32 H-tiles (N=32) = 256 blocks, 256 thr (8 waves)
// wave (g, half): gate g, output columns tile*32 + half*16 .. +15
// ------------------------------------------------------------------
__global__ __launch_bounds__(256) void lstm_scan(
    float* __restrict__ out, bf16_t* __restrict__ hbf,
    unsigned int* __restrict__ cnt,
    const bf16_t* __restrict__ xbf,
    const bf16_t* __restrict__ whbf, const bf16_t* __restrict__ wxbf,
    const float* __restrict__ bfv, const float* __restrict__ biv,
    const float* __restrict__ bcv, const float* __restrict__ bov)
{
    extern __shared__ char smem[];
    float* ldsBias  = (float*)(smem + LDS_BIAS_OFF);
    float* ldsGates = (float*)(smem + LDS_GATE_OFF);
    float* ldsC     = (float*)(smem + LDS_C_OFF);

    const int tid  = threadIdx.x;
    const int lane = tid & 31, wid = tid >> 5;
    const int hi   = lane >> 4, lrow = lane & 15;
    const int g    = wid >> 1,  half = wid & 1;
    const int chunk = blockIdx.x >> 5, tile = blockIdx.x & 31;
    const int nl    = half * 16 + lrow;           // 0..31 inside tile (also WMMA N for this lane)
    const int nglob = tile * 32 + nl;
    const int brow  = chunk * 16 + lrow;          // batch row carried by this lane (WMMA M)

    // ---- TDM: pull this block's W_h slice (4 x [32 x 1024] bf16) into LDS, padded ----
    if (wid == 0) {
        const unsigned int ldsBase = (unsigned int)(uintptr_t)(void*)smem;
        for (int gg = 0; gg < 4; ++gg) {
            uint64_t ga = (uint64_t)(uintptr_t)(whbf + ((size_t)gg * HID + tile * 32) * 1024);
            u32x4 d0;
            d0[0] = 1u;                                        // count=1, user descriptor
            d0[1] = ldsBase + (unsigned)(gg * 32 * LDS_WH_ROW);// lds_addr
            d0[2] = (unsigned int)ga;                          // global_addr[31:0]
            d0[3] = ((unsigned int)(ga >> 32) & 0x1FFFFFFu) | (2u << 30); // addr[56:32] | type=2
            i32x8 d1;
            // data_size=2B, pad_enable, pad_interval=7 (256 dw = 1024B), pad_amount=3 (4 dw = 16B)
            d1[0] = (int)((1u << 16) | (1u << 20) | (7u << 22) | (3u << 25));
            d1[1] = (int)((1024u & 0xFFFFu) << 16);            // tensor_dim0 lo16
            d1[2] = (int)((1024u >> 16) | (32u << 16));        // tensor_dim0 hi16 | tensor_dim1 lo16
            d1[3] = (int)(1024u << 16);                        // tensor_dim1 hi16 | tile_dim0
            d1[4] = (int)32u;                                  // tile_dim1 | tile_dim2(0)
            d1[5] = 1024;                                      // tensor_dim0_stride lo32
            d1[6] = 0;                                         // stride hi16 | dim1_stride lo16
            d1[7] = 0;
            i32x4 z4 = {0, 0, 0, 0};
            i32x8 z8 = {0, 0, 0, 0, 0, 0, 0, 0};
            __builtin_amdgcn_tensor_load_to_lds(d0, d1, z4, z4, z8, 0);
        }
        __builtin_amdgcn_s_wait_tensorcnt(0);
    }

    if (tid < 128) {
        const float* bp = (tid < 32) ? bfv : (tid < 64) ? biv : (tid < 96) ? bcv : bov;
        ldsBias[tid] = bp[tile * 32 + (tid & 31)];
    }
    ldsC[tid] = 0.f; ldsC[tid + 256] = 0.f;       // c0 = 0 (16x32 tile)
    __syncthreads();

    // Per-lane base pointers (affine in kb inside the loops below).
    const bf16_t* aHrow = hbf + (size_t)brow * HID + hi * 8;                 // A, h part
    const char*   whB   = smem + (size_t)(g * 32 + nl) * LDS_WH_ROW + hi * 16; // B, h part (LDS)
    const bf16_t* wxRow = wxbf + ((size_t)g * HID + nglob) * 1024 + hi * 8; // B, x part (global)

    for (int t = 0; t < SEQ; ++t) {
        const bf16_t* aXrow = xbf + ((size_t)brow * SEQ + t) * IN + hi * 8;
        __builtin_prefetch(aXrow + IN, 0, 3);     // global_prefetch_b8: next step's x row

        v8f acc = {0.f, 0.f, 0.f, 0.f, 0.f, 0.f, 0.f, 0.f};

        // --- h-part, LDS segment 0 (K = 0..511): B byte offset = k*2 + hi*16 ---
        #pragma unroll 4
        for (int kb = 0; kb < 16; ++kb) {
            const int k0 = kb * 32;
            union { v16bf v; v8bf h[2]; } Am, Bm;
            const bf16_t* pa = aHrow + k0;
            Am.h[0] = *(const v8bf*)pa;
            Am.h[1] = *(const v8bf*)(pa + 16);
            const char* pb = whB + k0 * 2;
            Bm.h[0] = *(const v8bf*)pb;
            Bm.h[1] = *(const v8bf*)(pb + 32);
            acc = __builtin_amdgcn_wmma_f32_16x16x32_bf16(
                false, Am.v, false, Bm.v, (short)0, acc, false, false);
        }
        // --- h-part, LDS segment 1 (K = 512..1023): +16B TDM pad, constant-folded ---
        #pragma unroll 4
        for (int kb = 16; kb < 32; ++kb) {
            const int k0 = kb * 32;
            union { v16bf v; v8bf h[2]; } Am, Bm;
            const bf16_t* pa = aHrow + k0;
            Am.h[0] = *(const v8bf*)pa;
            Am.h[1] = *(const v8bf*)(pa + 16);
            const char* pb = whB + k0 * 2 + 16;
            Bm.h[0] = *(const v8bf*)pb;
            Bm.h[1] = *(const v8bf*)(pb + 32);
            acc = __builtin_amdgcn_wmma_f32_16x16x32_bf16(
                false, Am.v, false, Bm.v, (short)0, acc, false, false);
        }
        // --- x-part (K = 1024..2047): A and B both from global ---
        #pragma unroll 4
        for (int kb = 0; kb < 32; ++kb) {
            const int kk = kb * 32;
            union { v16bf v; v8bf h[2]; } Am, Bm;
            const bf16_t* pa = aXrow + kk;
            Am.h[0] = *(const v8bf*)pa;
            Am.h[1] = *(const v8bf*)(pa + 16);
            const bf16_t* pb = wxRow + kk;
            Bm.h[0] = *(const v8bf*)pb;
            Bm.h[1] = *(const v8bf*)(pb + 16);
            acc = __builtin_amdgcn_wmma_f32_16x16x32_bf16(
                false, Am.v, false, Bm.v, (short)0, acc, false, false);
        }

        // scatter gate tile to LDS (lane = N, vgpr r = M = r + 8*hi), + bias
        const float bv = ldsBias[g * 32 + nl];
        #pragma unroll
        for (int r = 0; r < 8; ++r)
            ldsGates[(g * 16 + r + hi * 8) * 32 + nl] = acc[r] + bv;
        __syncthreads();

        // elementwise LSTM cell: 512 (row,col) pairs, 2 per thread
        #pragma unroll
        for (int e = 0; e < 2; ++e) {
            const int idx = tid + e * 256;
            const int row = idx >> 5, col = idx & 31;
            const float fg = sigm (ldsGates[(0 * 16 + row) * 32 + col]);
            const float ig = sigm (ldsGates[(1 * 16 + row) * 32 + col]);
            const float gg = tanha(ldsGates[(2 * 16 + row) * 32 + col]);
            const float og = sigm (ldsGates[(3 * 16 + row) * 32 + col]);
            const float cn = ldsC[row * 32 + col] * fg + ig * gg;
            ldsC[row * 32 + col] = cn;
            const float hn = tanha(cn) * og;
            const int b = chunk * 16 + row, n = tile * 32 + col;
            out[((size_t)b * SEQ + t) * HID + n] = hn;
            hbf[(size_t)b * HID + n] = (bf16_t)hn;
        }

        // grid-wide generation barrier: everyone must see the new h before step t+1
        if (t != SEQ - 1) {
            __threadfence();
            __syncthreads();
            if (tid == 0) {
                __hip_atomic_fetch_add(&cnt[t], 1u, __ATOMIC_RELEASE, __HIP_MEMORY_SCOPE_AGENT);
                while (__hip_atomic_load(&cnt[t], __ATOMIC_ACQUIRE, __HIP_MEMORY_SCOPE_AGENT)
                       < gridDim.x)
                    __builtin_amdgcn_s_sleep(1);
            }
            __syncthreads();
        }
    }
}

// ------------------------------------------------------------------
extern "C" void kernel_launch(void* const* d_in, const int* in_sizes, int n_in,
                              void* d_out, int out_size, void* d_ws, size_t ws_size,
                              hipStream_t stream)
{
    const float* x   = (const float*)d_in[0];
    const float* Wf  = (const float*)d_in[1]; const float* bf_ = (const float*)d_in[2];
    const float* Wi  = (const float*)d_in[3]; const float* bi_ = (const float*)d_in[4];
    const float* Wc  = (const float*)d_in[5]; const float* bc_ = (const float*)d_in[6];
    const float* Wo  = (const float*)d_in[7]; const float* bo_ = (const float*)d_in[8];

    char* ws = (char*)d_ws;
    bf16_t*       wh  = (bf16_t*)(ws + OFF_WH);
    bf16_t*       wx  = (bf16_t*)(ws + OFF_WX);
    bf16_t*       xbf = (bf16_t*)(ws + OFF_XBF);
    bf16_t*       hbf = (bf16_t*)(ws + OFF_HBF);
    unsigned int* cnt = (unsigned int*)(ws + OFF_CNT);

    pack_weights<<<4096, 256, 0, stream>>>(Wf, Wi, Wc, Wo, wh, wx);
    pack_x      <<<8192, 256, 0, stream>>>(x, xbf);
    init_state  <<<256,  256, 0, stream>>>((unsigned int*)hbf, cnt);
    lstm_scan   <<<256,  256, SMEM_BYTES, stream>>>(
        (float*)d_out, hbf, cnt, xbf, wh, wx, bf_, bi_, bc_, bo_);
}
